// GCN_66211215835752
// MI455X (gfx1250) — compile-verified, matrix operand order checked by reading, not run on hardware
//
#include <hip/hip_runtime.h>
#include <hip/hip_bf16.h>
#include <math.h>

// ---------------------------------------------------------------------------
// Types for CDNA5 WMMA (wave32): D(16x16 f32) = A(16x32 bf16) x B(32x16 bf16) + C
// ---------------------------------------------------------------------------
typedef __attribute__((ext_vector_type(16))) __bf16         v16bf;
typedef __attribute__((ext_vector_type(16))) unsigned short v16u;
typedef __attribute__((ext_vector_type(8)))  float          v8f;

#define HID 128   // output columns of both GEMMs (HID_CH == OUT_CH == 128)

// fp32 -> bf16 bits, round-to-nearest-even
static __device__ __forceinline__ unsigned int f2bf(float f) {
    unsigned int u = __float_as_uint(f);
    unsigned int r = u + 0x7FFFu + ((u >> 16) & 1u);
    return r >> 16;
}
static __device__ __forceinline__ unsigned int pack_bf16(float lo, float hi) {
    return f2bf(lo) | (f2bf(hi) << 16);
}

// K index inside a 16x32 bf16 A/B fragment (ISA 7.12.2): lane half (0/1), j in [0,16)
static __device__ __forceinline__ int frag_k(int half, int j) {
    return ((j >> 3) << 4) + (half << 3) + (j & 7);
}

// ---------------------------------------------------------------------------
// GEMM: H[M x 128] = act(X[M x K]) @ W[K x 128], bf16 WMMA, f32 accumulate.
// Block = 128 threads = 4 waves; block computes 64 rows x 128 cols;
// each wave owns 16 rows x 128 cols = 8 v8f accumulators.
// Tiles staged in LDS pre-swizzled to fragment layout; staged as bf16 pairs
// (float2 global load -> packed u32 -> single ds_store_b32).
// ---------------------------------------------------------------------------
template <int K, bool RELU>
__global__ __launch_bounds__(128)
void gcn_gemm_wmma(const float* __restrict__ X, const float* __restrict__ W,
                   float* __restrict__ H, int M)
{
    __shared__ unsigned short As[4 * 32 * 16]; // [wave][lane][16]  (4 KB)
    __shared__ unsigned short Bs[8 * 32 * 16]; // [ntile][lane][16] (8 KB)

    const int tid      = threadIdx.x;
    const int wave     = tid >> 5;
    const int lane     = tid & 31;
    const int half     = lane >> 4;
    const int lane16   = lane & 15;
    const int row_base = blockIdx.x * 64;

    v8f acc[8] = {};

    constexpr int KSTEPS = K / 32;
#pragma unroll 1
    for (int kb = 0; kb < KSTEPS; ++kb) {
        const int k0 = kb * 32;

        // ---- stage A tile (64 rows x 32 k) as bf16 pairs, fragment-ready ----
#pragma unroll
        for (int t = 0; t < 8; ++t) {            // 1024 pairs / 128 threads
            int i    = (tid + t * 128) << 1;     // even element index
            int w    = i >> 9;
            int rem  = i & 511;
            int ln   = rem >> 4;
            int j    = rem & 15;                 // even
            int m    = (w << 4) | (ln & 15);
            int k    = frag_k(ln >> 4, j);       // even; k,k+1 contiguous
            int row  = row_base + m;
            row      = row < M ? row : M - 1;    // clamp (stores masked later)
            float2 v = *(const float2*)&X[(size_t)row * K + k0 + k];
            if (RELU) { v.x = fmaxf(v.x, 0.0f); v.y = fmaxf(v.y, 0.0f); }
            ((unsigned int*)As)[i >> 1] = pack_bf16(v.x, v.y);
        }
        // ---- stage B tile (32 k x 128 n) as bf16 pairs, fragment-ready ----
#pragma unroll
        for (int t = 0; t < 16; ++t) {           // 2048 pairs / 128 threads
            int i    = (tid + t * 128) << 1;
            int nt   = i >> 9;
            int rem  = i & 511;
            int ln   = rem >> 4;
            int j    = rem & 15;                 // even
            int n    = (nt << 4) | (ln & 15);
            int k    = frag_k(ln >> 4, j);
            float w0 = W[(size_t)(k0 + k)     * HID + n];
            float w1 = W[(size_t)(k0 + k + 1) * HID + n];
            ((unsigned int*)Bs)[i >> 1] = pack_bf16(w0, w1);
        }
        __syncthreads();

        // ---- WMMA: one A fragment per wave, 8 B fragments (n tiles) ----
        v16u au = *(const v16u*)&As[((wave << 5) | lane) << 4];
        v16bf av = __builtin_bit_cast(v16bf, au);
#pragma unroll
        for (int nt = 0; nt < 8; ++nt) {
            v16u bu = *(const v16u*)&Bs[((nt << 5) | lane) << 4];
            v16bf bv = __builtin_bit_cast(v16bf, bu);
            acc[nt] = __builtin_amdgcn_wmma_f32_16x16x32_bf16(
                false, av, false, bv, (short)0, acc[nt], false, false);
        }
        __syncthreads();
    }

    // ---- epilogue: C/D layout -> H[row, col] (ISA 7.12.2 32-bit C/D) ----
    const int m_base = row_base + (wave << 4) + (half << 3);
    if (row_base + 64 <= M) {                    // fast path: whole block valid
#pragma unroll
        for (int nt = 0; nt < 8; ++nt)
#pragma unroll
            for (int j = 0; j < 8; ++j)
                H[(size_t)(m_base + j) * HID + (nt << 4) + lane16] = acc[nt][j];
    } else {
#pragma unroll
        for (int nt = 0; nt < 8; ++nt)
#pragma unroll
            for (int j = 0; j < 8; ++j)
                if (m_base + j < M)
                    H[(size_t)(m_base + j) * HID + (nt << 4) + lane16] = acc[nt][j];
    }
}

// ---------------------------------------------------------------------------
// Degree kernels (self-loops folded in as deg init = 1.0)
// ---------------------------------------------------------------------------
__global__ void deg_init_k(float* __restrict__ deg, int n) {
    int i = blockIdx.x * blockDim.x + threadIdx.x;
    if (i < n) deg[i] = 1.0f;
}
__global__ void deg_count_k(const int* __restrict__ dst, float* __restrict__ deg, int E) {
    int e = blockIdx.x * blockDim.x + threadIdx.x;
    if (e < E) atomicAdd(&deg[dst[e]], 1.0f);
}
__global__ void deg_rsqrt_k(float* __restrict__ deg, int n) {
    int i = blockIdx.x * blockDim.x + threadIdx.x;
    if (i < n) deg[i] = rsqrtf(deg[i]);
}

// ---------------------------------------------------------------------------
// out[i,c] = bias[c] + h[i,c] * dis[i]^2   (bias + fused self-loop), float4
// ---------------------------------------------------------------------------
__global__ __launch_bounds__(256)
void self_init_k(const float* __restrict__ h, const float* __restrict__ dis,
                 const float* __restrict__ bias, float* __restrict__ out, int n)
{
    int i = blockIdx.x * blockDim.x + threadIdx.x;   // over n * 32 float4s
    if (i >= n * (HID / 4)) return;
    int node = i >> 5, c4 = i & 31;
    float d2 = dis[node]; d2 *= d2;
    float4 hv = ((const float4*)h)[i];
    float4 bv = ((const float4*)bias)[c4];
    float4 o = { bv.x + hv.x * d2, bv.y + hv.y * d2,
                 bv.z + hv.z * d2, bv.w + hv.w * d2 };
    ((float4*)out)[i] = o;
}

// ---------------------------------------------------------------------------
// Edge scatter: one edge per wave. e is wave-uniform (readfirstlane) so
// src/dst/deg_isqrt loads become scalar (SMEM); gather h[s,:] as float4/lane
// (512B coalesced per wave), then 4 x global_atomic_add_f32.
// ---------------------------------------------------------------------------
__global__ __launch_bounds__(256)
void scatter_k(const int* __restrict__ src, const int* __restrict__ dst,
               const float* __restrict__ dis, const float* __restrict__ h,
               float* __restrict__ out, int E)
{
    int wave = threadIdx.x >> 5;
    int lane = threadIdx.x & 31;
    int e = __builtin_amdgcn_readfirstlane(blockIdx.x * 8 + wave);
    if (e >= E) return;
    int s = src[e], d = dst[e];
    float nrm = dis[s] * dis[d];
    float4 v = ((const float4*)(h + (size_t)s * HID))[lane];
    float* op = out + (size_t)d * HID + lane * 4;
    atomicAdd(op + 0, v.x * nrm);
    atomicAdd(op + 1, v.y * nrm);
    atomicAdd(op + 2, v.z * nrm);
    atomicAdd(op + 3, v.w * nrm);
}

// ---------------------------------------------------------------------------
// Orchestration
// ---------------------------------------------------------------------------
extern "C" void kernel_launch(void* const* d_in, const int* in_sizes, int n_in,
                              void* d_out, int out_size, void* d_ws, size_t ws_size,
                              hipStream_t stream)
{
    const float* x  = (const float*)d_in[0];
    const int*   ei = (const int*)  d_in[1];
    const float* W1 = (const float*)d_in[2];
    const float* b1 = (const float*)d_in[3];
    const float* W2 = (const float*)d_in[4];
    const float* b2 = (const float*)d_in[5];

    const int IN_CH = 512;
    const int N = in_sizes[0] / IN_CH;   // 100000
    const int E = in_sizes[1] / 2;       // 1600000
    const int* src = ei;
    const int* dst = ei + E;
    float* out = (float*)d_out;

    // workspace layout
    char* ws = (char*)d_ws;
    float* deg  = (float*)ws;                                   // N (deg -> deg_isqrt in place)
    size_t o1 = ((size_t)N * 4 + 511) & ~(size_t)511;
    float* bufA = (float*)(ws + o1);                            // N*128 (h1, later h2)
    size_t o2 = o1 + (((size_t)N * HID * 4 + 511) & ~(size_t)511);
    float* bufB = (float*)(ws + o2);                            // N*128 (agg1)

    const int T = 256;
    // 1) degrees (with self-loop) -> rsqrt
    deg_init_k <<<(N + T - 1) / T, T, 0, stream>>>(deg, N);
    deg_count_k<<<(E + T - 1) / T, T, 0, stream>>>(dst, deg, E);
    deg_rsqrt_k<<<(N + T - 1) / T, T, 0, stream>>>(deg, N);

    // 2) h1 = x @ W1  (bf16 WMMA)
    gcn_gemm_wmma<512, false><<<(N + 63) / 64, 128, 0, stream>>>(x, W1, bufA, N);

    // 3) agg1 = b1 + h1 * dis^2 (self loop), then scatter edges
    self_init_k<<<((size_t)N * (HID / 4) + T - 1) / T, T, 0, stream>>>(bufA, deg, b1, bufB, N);
    scatter_k  <<<(E + 7) / 8, T, 0, stream>>>(src, dst, deg, bufA, bufB, E);

    // 4) h2 = relu(agg1) @ W2  (ReLU fused into A-tile load); bufA dead -> reuse
    gcn_gemm_wmma<128, true><<<(N + 63) / 64, 128, 0, stream>>>(bufB, W2, bufA, N);

    // 5) out = b2 + h2 * dis^2, then scatter edges into d_out
    self_init_k<<<((size_t)N * (HID / 4) + T - 1) / T, T, 0, stream>>>(bufA, deg, b2, out, N);
    scatter_k  <<<(E + 7) / 8, T, 0, stream>>>(src, dst, deg, bufA, out, E);
}
